// AttentionRNN_67851893342902
// MI455X (gfx1250) — matrix operand order repaired and checked by reading, not compiled, not run on hardware
//
#include <hip/hip_runtime.h>

#define B_   64
#define S_   512
#define NH_  4
#define DIN_ 256
#define H_   256
#define AH_  256
#define OUT_ 3

#define LDSPAD 264   // 256 + 8 bf16 pad: lane half maps to banks 4*lc mod 64 -> conflict-free

typedef __attribute__((ext_vector_type(16))) __bf16          v16bf;
typedef __attribute__((ext_vector_type(16))) unsigned short  v16u;
typedef __attribute__((ext_vector_type(8)))  float           v8f;

union Frag16 { v16u u; v16bf b; };

__device__ __forceinline__ unsigned short f2bs(float f) {
  unsigned u = __float_as_uint(f);
  u += 0x7fffu + ((u >> 16) & 1u);          // round-to-nearest-even to bf16
  return (unsigned short)(u >> 16);
}
__device__ __forceinline__ float bs2f(unsigned short s) {
  return __uint_as_float(((unsigned)s) << 16);
}

// ---------------------------------------------------------------------------
// Wave-level 256x256 GEMM tile worker (32 waves cover the full 16x16 tile grid)
// A: [256][KTOT] bf16 row-major  (or CONCAT: [x_t fp32 | attnp bf16], KTOT=512)
// BT: B^T with row stride LDB, bf16; may live in LDS (LDB=LDSPAD, ds_load
// fragments, ~30cy) or global (LDB=KTOT, L2).  PF: prefetch next k-step's B.
// ---------------------------------------------------------------------------
template<int KTOT, int LDB, bool CONCAT, bool RELU, bool F32OUT, bool PF>
__device__ __forceinline__ void wave_gemm_256(
    int wave, int lane, int t,
    const unsigned short* __restrict__ A,
    const float*          __restrict__ X,
    const unsigned short* __restrict__ A2,
    const unsigned short* __restrict__ BT,
    const float*          __restrict__ bias,
    float*                __restrict__ outF,
    unsigned short*       __restrict__ outB)
{
  const int half = lane >> 4;
  const int lc   = lane & 15;
  const int mt   = wave >> 1;          // 16 M-tiles
  const int nb   = (wave & 1) * 8;     // 8 N-tiles per wave
  const int row  = mt * 16 + lc;       // A-matrix row for this lane

  v8f acc[8];
  #pragma unroll
  for (int i = 0; i < 8; ++i)
    #pragma unroll
    for (int r = 0; r < 8; ++r) acc[i][r] = 0.f;

  const float*          xrow  = nullptr;
  const unsigned short* a2row = nullptr;
  const unsigned short* arow  = nullptr;
  if (CONCAT) {
    int b = row >> 2, hh = row & 3;
    xrow  = X + ((size_t)((b * S_ + t) * NH_ + hh)) * DIN_;
    a2row = A2 + (size_t)row * AH_;
  } else {
    arow = A + (size_t)row * KTOT;
  }

  const unsigned short* brow = BT + (size_t)(nb * 16 + lc) * LDB + half * 8;

  for (int k0 = 0; k0 < KTOT; k0 += 32) {
    if (PF && (k0 + 32 < KTOT)) {
      #pragma unroll
      for (int i = 0; i < 8; ++i)
        __builtin_prefetch(brow + (size_t)i * 16 * LDB + k0 + 32, 0, 0);
    }
    // A fragment: elems 0..7 -> K = k0+half*8+j ; elems 8..15 -> K = k0+16+half*8+j
    Frag16 a;
    if (CONCAT) {
      if (k0 < DIN_) {
        const float* p = xrow + k0 + half * 8;
        #pragma unroll
        for (int j = 0; j < 8; ++j) { a.u[j] = f2bs(p[j]); a.u[j + 8] = f2bs(p[16 + j]); }
      } else {
        const unsigned short* p = a2row + (k0 - DIN_) + half * 8;
        #pragma unroll
        for (int j = 0; j < 8; ++j) { a.u[j] = p[j]; a.u[j + 8] = p[16 + j]; }
      }
    } else {
      const unsigned short* p = arow + k0 + half * 8;
      #pragma unroll
      for (int j = 0; j < 8; ++j) { a.u[j] = p[j]; a.u[j + 8] = p[16 + j]; }
    }

    // 8 B fragments (col N = lc, elems 0..7 -> K = k0+half*8+j ; elems 8..15 -> K = k0+16+half*8+j)
    #pragma unroll
    for (int i = 0; i < 8; ++i) {
      const unsigned short* q = brow + (size_t)i * 16 * LDB + k0;
      Frag16 bf;
      #pragma unroll
      for (int j = 0; j < 16; ++j) bf.u[j] = q[j < 8 ? j : j + 8];
      acc[i] = __builtin_amdgcn_wmma_f32_16x16x32_bf16(
          false, a.b, false, bf.b, (short)0, acc[i], false, false);
    }
  }

  #pragma unroll
  for (int i = 0; i < 8; ++i) {
    int n = (nb + i) * 16 + lc;
    float bv = bias[n];
    #pragma unroll
    for (int r = 0; r < 8; ++r) {
      int m = mt * 16 + half * 8 + r;       // D row = half*8 + r (ISA C/D layout)
      float v = acc[i][r] + bv;
      if (RELU) v = fmaxf(v, 0.f);
      if (F32OUT) outF[(size_t)m * 256 + n] = v;
      else        outB[(size_t)m * 256 + n] = f2bs(v);
    }
  }
}

// ---------------------------------------------------------------------------
// Pre-pass kernels (fully parallel, off the serial path)
// ---------------------------------------------------------------------------
__global__ void zero_kernel(float* p, int n) {
  int i = blockIdx.x * blockDim.x + threadIdx.x;
  if (i < n) p[i] = 0.f;
}

// WT[n*K + k] = bf16(W[k*N + n])
__global__ void wt_kernel(const float* __restrict__ W, unsigned short* __restrict__ WT,
                          int K, int N) {
  int o = blockIdx.x * blockDim.x + threadIdx.x;
  if (o >= K * N) return;
  int n = o / K, k = o % K;
  WT[o] = f2bs(W[(size_t)k * N + n]);
}

// u = Wq @ w2 ; c = bq . w2   (folds the qx GEMM off the serial path)
__global__ void u_kernel(const float* __restrict__ Wq, const float* __restrict__ wmlp,
                         const float* __restrict__ bq, float* __restrict__ u,
                         float* __restrict__ c) {
  int k = threadIdx.x;
  float s = 0.f;
  for (int n = 0; n < H_; ++n) s += Wq[k * H_ + n] * wmlp[H_ + n];
  u[k] = s;
  if (k == 0) {
    float cc = 0.f;
    for (int n = 0; n < H_; ++n) cc += bq[n] * wmlp[H_ + n];
    *c = cc;
  }
}

// kx[t,b,i,:] = x[b,t,i,:] @ Wk + bk  for all 524288 rows (stored bf16)
__global__ __launch_bounds__(256) void kx_kernel(
    const float* __restrict__ x, const unsigned short* __restrict__ WkT,
    const float* __restrict__ bk, unsigned short* __restrict__ kx) {
  int tid = threadIdx.x, lane = tid & 31, wave = tid >> 5;
  int half = lane >> 4, lc = lane & 15;
  int mt = blockIdx.x * 4 + (wave >> 1);
  int nb = (wave & 1) * 8;
  int row = mt * 16 + lc;                      // rho = t*256 + b*4 + i
  int t = row >> 8, rm = row & 255, b = rm >> 2, hh = rm & 3;
  const float* xrow = x + ((size_t)((b * S_ + t) * NH_ + hh)) * DIN_;
  const unsigned short* brow = WkT + (size_t)(nb * 16 + lc) * DIN_ + half * 8;

  v8f acc[8];
  #pragma unroll
  for (int i = 0; i < 8; ++i)
    #pragma unroll
    for (int r = 0; r < 8; ++r) acc[i][r] = 0.f;

  for (int k0 = 0; k0 < DIN_; k0 += 32) {
    Frag16 a;
    const float* p = xrow + k0 + half * 8;
    #pragma unroll
    for (int j = 0; j < 8; ++j) { a.u[j] = f2bs(p[j]); a.u[j + 8] = f2bs(p[16 + j]); }

    #pragma unroll
    for (int i = 0; i < 8; ++i) {
      const unsigned short* q = brow + (size_t)i * 16 * DIN_ + k0;
      Frag16 bf;
      #pragma unroll
      for (int j = 0; j < 16; ++j) bf.u[j] = q[j < 8 ? j : j + 8];
      acc[i] = __builtin_amdgcn_wmma_f32_16x16x32_bf16(
          false, a.b, false, bf.b, (short)0, acc[i], false, false);
    }
  }
  #pragma unroll
  for (int i = 0; i < 8; ++i) {
    int n = (nb + i) * 16 + lc;
    float bv = bk[n];
    #pragma unroll
    for (int r = 0; r < 8; ++r) {
      int m = mt * 16 + half * 8 + r;
      kx[(size_t)m * H_ + n] = f2bs(acc[i][r] + bv);
    }
  }
}

// kw[rho] = kx[rho,:] . w1   (one wave per row, shuffle reduce)
__global__ __launch_bounds__(256) void kw_kernel(
    const unsigned short* __restrict__ kx, const float* __restrict__ wmlp,
    float* __restrict__ kw) {
  int tid = threadIdx.x, lane = tid & 31, wave = tid >> 5;
  size_t row = (size_t)blockIdx.x * 8 + wave;
  const unsigned short* kr = kx + row * H_;
  int kb = lane * 8;
  float p = 0.f;
  #pragma unroll
  for (int j = 0; j < 8; ++j) p += bs2f(kr[kb + j]) * wmlp[kb + j];
  #pragma unroll
  for (int off = 16; off > 0; off >>= 1) p += __shfl_down(p, off);
  if (lane == 0) kw[row] = p;
}

// ---------------------------------------------------------------------------
// Persistent sequential scan: 1 workgroup, 1024 threads (32 wave32, one WGP).
// Wproj^T and W2^T live in LDS (2 x 132KB, bank-conflict-free padded rows),
// loaded once; W1^T (256KB, K=512) streams from L2 with prefetch.
// ---------------------------------------------------------------------------
__global__ __launch_bounds__(1024) void rnn_kernel(
    const float* __restrict__ x,
    const float* __restrict__ bproj, const float* __restrict__ b1,
    const float* __restrict__ b2,
    const float* __restrict__ Wo, const float* __restrict__ bo,
    const unsigned short* __restrict__ WpT, const unsigned short* __restrict__ W1T,
    const unsigned short* __restrict__ W2T,
    const unsigned short* __restrict__ kx, const float* __restrict__ kw,
    const float* __restrict__ u, const float* __restrict__ cptr,
    unsigned short* __restrict__ attnb, unsigned short* __restrict__ attnpb,
    unsigned short* __restrict__ zb,
    float* __restrict__ out)   // out[0..191]=log_softmax(last), out+192 = h_final
{
  __shared__ unsigned short sWp[256 * LDSPAD];   // 132 KB
  __shared__ unsigned short sW2[256 * LDSPAD];   // 132 KB
  __shared__ float sU[256];
  __shared__ float sQw[256];
  __shared__ float sS[256 * 4];
  __shared__ float sLog[192];
  __shared__ float sC;

  int tid = threadIdx.x, lane = tid & 31, wave = tid >> 5;
  float* Hst = out + B_ * OUT_;                 // recurrent hidden, fp32, L2-resident

  // One-time LDS residency for the two K=256 weight matrices (coalesced).
  for (int idx = tid; idx < 256 * 256; idx += 1024) {
    int r = idx >> 8, cx = idx & 255;
    sWp[r * LDSPAD + cx] = WpT[idx];
    sW2[r * LDSPAD + cx] = W2T[idx];
  }
  if (tid < 256) sU[tid] = u[tid];
  if (tid == 0)  sC = *cptr;
  __syncthreads();

  for (int t = 0; t < S_; ++t) {
    // (1) qw[m] = hidden[m,:] . u + c      (4 lanes per row, coalesced, shfl-reduce)
    {
      int m = tid >> 2, k0 = tid & 3;
      const float* hr = Hst + (size_t)m * H_;
      float p = 0.f;
      for (int k = k0; k < H_; k += 4) p += hr[k] * sU[k];
      p += __shfl_down(p, 2);
      p += __shfl_down(p, 1);
      if (k0 == 0) sQw[m] = p;
    }
    __syncthreads();

    // (2) score s[m][j] = softmax_j(tanh(qw[m] + kw[t,b,j]))
    if (tid < 256) {
      int m = tid, b = m >> 2;
      float qv = sQw[m] + sC;
      const float* kr = kw + (size_t)t * 256 + b * 4;
      float e0 = tanhf(qv + kr[0]), e1 = tanhf(qv + kr[1]);
      float e2 = tanhf(qv + kr[2]), e3 = tanhf(qv + kr[3]);
      float mx = fmaxf(fmaxf(e0, e1), fmaxf(e2, e3));
      e0 = __expf(e0 - mx); e1 = __expf(e1 - mx);
      e2 = __expf(e2 - mx); e3 = __expf(e3 - mx);
      float inv = 1.f / (e0 + e1 + e2 + e3);
      sS[m * 4 + 0] = e0 * inv; sS[m * 4 + 1] = e1 * inv;
      sS[m * 4 + 2] = e2 * inv; sS[m * 4 + 3] = e3 * inv;
    }
    __syncthreads();

    // (3) attn[m][h] = sum_j s[m][j] * kx[t,b,j,h]  -> bf16 staging
    for (int p = tid; p < B_ * H_; p += 1024) {
      int b = p >> 8, h = p & 255;
      const unsigned short* kp = kx + (((size_t)t * 256 + b * 4)) * H_ + h;
      float k0v = bs2f(kp[0]),      k1v = bs2f(kp[256]);
      float k2v = bs2f(kp[512]),    k3v = bs2f(kp[768]);
      #pragma unroll
      for (int i = 0; i < 4; ++i) {
        const float* sp = &sS[(b * 4 + i) * 4];
        float v = sp[0] * k0v + sp[1] * k1v + sp[2] * k2v + sp[3] * k3v;
        attnb[(size_t)(b * 4 + i) * H_ + h] = f2bs(v);
      }
    }
    __threadfence(); __syncthreads();

    // (4) attnp = attn @ Wproj + bproj          (B from LDS)
    wave_gemm_256<256, LDSPAD, false, false, false, false>(
        wave, lane, t, attnb, nullptr, nullptr, sWp, bproj, nullptr, attnpb);
    __threadfence(); __syncthreads();

    // (5) z = relu([x_t | attnp] @ W1 + b1)     (K = 512, B from L2 + prefetch)
    wave_gemm_256<512, 512, true, true, false, true>(
        wave, lane, t, nullptr, x, attnpb, W1T, b1, nullptr, zb);
    __threadfence(); __syncthreads();

    // (6) hidden = z @ W2 + b2                  (B from LDS, fp32 state out)
    wave_gemm_256<256, LDSPAD, false, false, true, false>(
        wave, lane, t, zb, nullptr, nullptr, sW2, b2, Hst, nullptr);
    __threadfence(); __syncthreads();
  }

  // logits + log_softmax for the final step only
  if (tid < B_ * OUT_) {
    int b = tid / OUT_, o = tid % OUT_;
    const float* hb = Hst + (size_t)b * (NH_ * H_);
    float a = bo[o];
    for (int q = 0; q < NH_ * H_; ++q) a += hb[q] * Wo[q * OUT_ + o];
    sLog[tid] = a;
  }
  __syncthreads();
  if (tid < B_ * OUT_) {
    int b = tid / OUT_;
    float l0 = sLog[b * 3], l1 = sLog[b * 3 + 1], l2 = sLog[b * 3 + 2];
    float mx = fmaxf(l0, fmaxf(l1, l2));
    float lse = mx + logf(__expf(l0 - mx) + __expf(l1 - mx) + __expf(l2 - mx));
    out[tid] = sLog[tid] - lse;
  }
}

// ---------------------------------------------------------------------------
extern "C" void kernel_launch(void* const* d_in, const int* in_sizes, int n_in,
                              void* d_out, int out_size, void* d_ws, size_t ws_size,
                              hipStream_t stream) {
  const float* x     = (const float*)d_in[0];
  const float* Wk    = (const float*)d_in[1];
  const float* bk    = (const float*)d_in[2];
  const float* Wq    = (const float*)d_in[3];
  const float* bq    = (const float*)d_in[4];
  const float* wmlp  = (const float*)d_in[5];
  const float* Wproj = (const float*)d_in[6];
  const float* bproj = (const float*)d_in[7];
  const float* W1    = (const float*)d_in[8];
  const float* b1    = (const float*)d_in[9];
  const float* W2    = (const float*)d_in[10];
  const float* b2    = (const float*)d_in[11];
  const float* Wo    = (const float*)d_in[12];
  const float* bo    = (const float*)d_in[13];
  float* out = (float*)d_out;
  char*  ws  = (char*)d_ws;

  // Workspace layout (256B-aligned): ~66 MB total
  unsigned short* WkT    = (unsigned short*)(ws + 0);         // 128 KB
  unsigned short* WpT    = (unsigned short*)(ws + 131072);    // 128 KB
  unsigned short* W1T    = (unsigned short*)(ws + 262144);    // 256 KB
  unsigned short* W2T    = (unsigned short*)(ws + 524288);    // 128 KB
  float*          ubuf   = (float*)(ws + 655360);             // 1 KB
  float*          cbuf   = (float*)(ws + 656384);
  unsigned short* attnb  = (unsigned short*)(ws + 786432);    // 128 KB
  unsigned short* attnpb = (unsigned short*)(ws + 917504);    // 128 KB
  float*          kwbuf  = (float*)(ws + 1048576);            // 512 KB
  unsigned short* zbuf   = (unsigned short*)(ws + 1572864);   // 128 KB
  unsigned short* kxbuf  = (unsigned short*)(ws + 1703936);   // 64 MB (bf16 kx_all)

  int nout = B_ * OUT_ + B_ * NH_ * H_;                       // 65728
  zero_kernel<<<(nout + 255) / 256, 256, 0, stream>>>(out, nout);

  wt_kernel<<<(256 * 256 + 255) / 256, 256, 0, stream>>>(Wk, WkT, 256, 256);
  wt_kernel<<<(256 * 256 + 255) / 256, 256, 0, stream>>>(Wproj, WpT, 256, 256);
  wt_kernel<<<(512 * 256 + 255) / 256, 256, 0, stream>>>(W1, W1T, 512, 256);
  wt_kernel<<<(256 * 256 + 255) / 256, 256, 0, stream>>>(W2, W2T, 256, 256);
  u_kernel<<<1, 256, 0, stream>>>(Wq, wmlp, bq, ubuf, cbuf);

  // 524288 rows / 16 per M-tile / 4 M-tiles per block
  kx_kernel<<<8192, 256, 0, stream>>>(x, WkT, bk, kxbuf);
  kw_kernel<<<65536, 256, 0, stream>>>(kxbuf, wmlp, kwbuf);

  rnn_kernel<<<1, 1024, 0, stream>>>(x, bproj, b1, b2, Wo, bo,
                                     WpT, W1T, W2T, kxbuf, kwbuf, ubuf, cbuf,
                                     attnb, attnpb, zbuf, out);
}
